// multi_head_attention_23338852286663
// MI455X (gfx1250) — matrix operand order repaired and checked by reading, not compiled, not run on hardware
//
#include <hip/hip_runtime.h>
#include <hip/hip_bf16.h>

// ---- CDNA5 WMMA vector types ----
typedef __attribute__((ext_vector_type(16))) __bf16 v16bf;
typedef __attribute__((ext_vector_type(2)))  __bf16 v2bf;
typedef __attribute__((ext_vector_type(8)))  float  v8f;
typedef __attribute__((ext_vector_type(4)))  unsigned int u32x4;

union ABf { v16bf v; u32x4 q[2]; unsigned int u[8]; unsigned short s[16]; };

__device__ __forceinline__ unsigned short f2bf(float f) {
    unsigned int u = __builtin_bit_cast(unsigned int, f);
    u = (u + 0x7fffu + ((u >> 16) & 1u)) >> 16;   // round-to-nearest-even
    return (unsigned short)u;
}

__device__ __forceinline__ unsigned int f2bf2(float lo, float hi) {
#if __has_builtin(__builtin_amdgcn_cvt_pk_bf16_f32)
    v2bf r = __builtin_amdgcn_cvt_pk_bf16_f32(lo, hi);
    return __builtin_bit_cast(unsigned int, r);
#else
    return (unsigned int)f2bf(lo) | ((unsigned int)f2bf(hi) << 16);
#endif
}

// Problem constants
#define NVr   300
#define Bb    32
#define Cc    64
#define NHh   64
#define Dd    512
#define Mm    1200
#define MPAD  1216    /* m padded to mult of 32 */

// =====================================================================
// Kernel 1 (templated): projection x@w (f32 -> bf16, WMMA bf16)
//   WHICH=0: Q -> Qb2[b][1200][512]   (row-major in d)
//   WHICH=1: K -> Kb2[b][300][512]    (x rows offset by 900)
//   WHICH=2: V -> Vb2[d][32][1216]    (m contiguous, packed b128 stores)
// grid (groups, 8 dg, 32 b), block 256 (8 waves; wave = one 16-m tile)
// =====================================================================
template<int WHICH>
__launch_bounds__(256, 2)
__global__ void proj_kernel(const float* __restrict__ x,
                            const float* __restrict__ w,
                            unsigned short* __restrict__ dst)
{
    __shared__ __align__(16) unsigned short wlds[64][72]; // [col][c], padded

    const int tid  = threadIdx.x;
    const int lane = tid & 31;
    const int wid  = tid >> 5;
    const int dg   = blockIdx.y;          // group of 64 output cols
    const int bbi  = blockIdx.z;          // batch
    const int hi   = (lane >> 4) & 1;
    const int nl   = lane & 15;
    const int kb8  = hi * 8;
    constexpr int R    = (WHICH == 1) ? NVr : Mm;
    constexpr int XOFF = (WHICH == 1) ? 900 : 0;

    // stage weight tile (64 c x 64 cols) transposed into LDS as bf16
    #pragma unroll
    for (int it = 0; it < 4; ++it) {
        const int idx = tid + it * 256;              // 0..1023
        const int c = idx >> 4, q4 = (idx & 15) * 4;
        const float4 wv4 = *(const float4*)(w + (size_t)c * Dd + dg * 64 + q4);
        wlds[q4 + 0][c] = f2bf(wv4.x);
        wlds[q4 + 1][c] = f2bf(wv4.y);
        wlds[q4 + 2][c] = f2bf(wv4.z);
        wlds[q4 + 3][c] = f2bf(wv4.w);
    }
    __syncthreads();

    const int mbase = (blockIdx.x * 8 + wid) * 16;   // row tile (m or n)
    if (mbase >= R) return;                          // after the only barrier

    // A fragments: 16 consecutive rows (fixed b), two K-steps of 32
    int mrow = mbase + nl; if (mrow > R - 1) mrow = R - 1;
    const float* xr = x + ((size_t)(XOFF + mrow) * Bb + bbi) * Cc;
    ABf A[2];
    #pragma unroll
    for (int s = 0; s < 2; ++s)
        #pragma unroll
        for (int h = 0; h < 2; ++h) {
            const float* p = xr + s * 32 + kb8 * 2 + h * 8;
            #pragma unroll
            for (int e = 0; e < 8; e += 2)
                A[s].u[(h * 8 + e) >> 1] = f2bf2(p[e], p[e + 1]);
        }

    v8f acc[4];
    #pragma unroll
    for (int ds = 0; ds < 4; ++ds) {
        v8f c = {};
        #pragma unroll
        for (int s = 0; s < 2; ++s) {
            ABf Bw;
            const unsigned short* wp = &wlds[ds * 16 + nl][s * 32 + hi * 16];
            Bw.q[0] = *(const u32x4*)(wp);
            Bw.q[1] = *(const u32x4*)(wp + 8);
            c = __builtin_amdgcn_wmma_f32_16x16x32_bf16(
                    false, A[s].v, false, Bw.v, (short)0, c, false, false);
        }
        acc[ds] = c;
    }

    if constexpr (WHICH == 2) {
        // V: lane's column d holds 8 consecutive m -> one packed b128 store
        #pragma unroll
        for (int ds = 0; ds < 4; ++ds) {
            const int d = dg * 64 + ds * 16 + nl;
            u32x4 pk = { f2bf2(acc[ds][0], acc[ds][1]),
                         f2bf2(acc[ds][2], acc[ds][3]),
                         f2bf2(acc[ds][4], acc[ds][5]),
                         f2bf2(acc[ds][6], acc[ds][7]) };
            *(u32x4*)(dst + ((size_t)d * Bb + bbi) * MPAD + mbase + hi * 8) = pk;
        }
    } else {
        // Q/K: transpose-bounce through per-wave LDS, coalesced b128 stores
        __shared__ __align__(16) unsigned short obuf[8][16][72];
        #pragma unroll
        for (int ds = 0; ds < 4; ++ds)
            #pragma unroll
            for (int r = 0; r < 8; ++r)
                obuf[wid][r + hi * 8][ds * 16 + nl] = f2bf(acc[ds][r]);
        // same-wave LDS is in-order; no block barrier needed
        const int rr = lane >> 1, half = lane & 1;
        const int mm = mbase + rr;
        if (mm < R) {
            const unsigned short* src = &obuf[wid][rr][half * 32];
            unsigned short* dp = dst + ((size_t)bbi * R + mm) * Dd + dg * 64 + half * 32;
            #pragma unroll
            for (int j = 0; j < 4; ++j)
                *(u32x4*)(dp + j * 8) = *(const u32x4*)(src + j * 8);
        }
    }
}

// =====================================================================
// Kernel 2: fused score -> softmax(over heads) -> o accumulation
// grid (19 n-tiles, 32 batch), block 256 (8 waves = 8 heads)
// Double-buffered LDS staging: only 2 barriers per m-iteration.
// =====================================================================
__launch_bounds__(256, 2)
__global__ void attn_kernel(const unsigned short* __restrict__ Qb2,
                            const unsigned short* __restrict__ Kb2,
                            const unsigned short* __restrict__ Vb2,
                            unsigned short* __restrict__ Ob)
{
    __shared__ __align__(16) float          scbuf[2][2][8][32][8];  // [buf][sub][r][lane][k]
    __shared__ __align__(16) unsigned short abuf[2][8][16][32];     // [buf][k][n][m2]

    const int tid  = threadIdx.x;
    const int lane = tid & 31;
    const int k    = tid >> 5;            // head = wave id
    const int nbase = blockIdx.x * 16;
    const int bbi   = blockIdx.y;
    const int hi  = (lane >> 4) & 1;
    const int nl  = lane & 15;
    const int kb8 = hi * 8;

    // persistent K^T B-tiles (B[32d x 16n]) for this head
    ABf KB[2];
    {
        int n = nbase + nl; if (n > NVr - 1) n = NVr - 1;
        const unsigned short* kr = Kb2 + ((size_t)bbi * NVr + n) * Dd + k * NHh;
        #pragma unroll
        for (int s = 0; s < 2; ++s) {
            const unsigned short* p = kr + s * 32 + hi * 16;
            KB[s].q[0] = *(const u32x4*)(p);
            KB[s].q[1] = *(const u32x4*)(p + 8);
        }
    }

    v8f oacc[4] = {};   // o strip: 16n x 64d (f32)

    for (int mi = 0; mi < 38; ++mi) {
        const int mbase = mi * 32;
        const int buf   = mi & 1;

        // prefetch next iteration's Q rows (global_prefetch_b8)
        if (mi + 1 < 38) {
            int mp = mbase + 32 + nl; if (mp > Mm - 1) mp = Mm - 1;
            __builtin_prefetch(Qb2 + ((size_t)bbi * Mm + mp) * Dd + k * NHh, 0, 3);
        }

        // ---- score: two 16-m subtiles, A=Q, B=K^T ----
        #pragma unroll
        for (int sub = 0; sub < 2; ++sub) {
            int m = mbase + sub * 16 + nl; if (m > Mm - 1) m = Mm - 1;
            const unsigned short* qr = Qb2 + ((size_t)bbi * Mm + m) * Dd + k * NHh;
            v8f c = {};
            #pragma unroll
            for (int s = 0; s < 2; ++s) {
                ABf Aq;
                const unsigned short* p = qr + s * 32 + hi * 16;
                Aq.q[0] = *(const u32x4*)(p);
                Aq.q[1] = *(const u32x4*)(p + 8);
                c = __builtin_amdgcn_wmma_f32_16x16x32_bf16(
                        false, Aq.v, false, KB[s].v, (short)0, c, false, false);
            }
            #pragma unroll
            for (int r = 0; r < 8; ++r)
                scbuf[buf][sub][r][lane][k] = c[r] * 0.125f;   // 1/sqrt(NH)
        }
        __syncthreads();   // barrier A: score tiles visible

        // ---- softmax over the 8 heads; 2 (m2,n) positions per thread ----
        #pragma unroll
        for (int qq = 0; qq < 2; ++qq) {
            const int p  = tid * 2 + qq;          // 0..511 = m2*16 + n
            const int m2 = p >> 4, n = p & 15;
            const int sub = m2 >> 4, mm = m2 & 15;
            const int r = mm & 7, h2 = mm >> 3;
            const float* sp = &scbuf[buf][sub][r][h2 * 16 + n][0];
            float v[8], mx = -1e30f;
            #pragma unroll
            for (int j = 0; j < 8; ++j) { v[j] = sp[j]; mx = fmaxf(mx, v[j]); }
            float sum = 0.f;
            #pragma unroll
            for (int j = 0; j < 8; ++j) { v[j] = __expf(v[j] - mx); sum += v[j]; }
            const float inv = (mbase + m2 < Mm) ? (1.f / sum) : 0.f; // zero m-tail
            #pragma unroll
            for (int j = 0; j < 8; ++j) abuf[buf][j][n][m2] = f2bf(v[j] * inv);
        }
        __syncthreads();   // barrier B: attn tiles visible

        // ---- o accumulation: A = attn^T (16n x 32m), B = V (32m x 16d) ----
        // (no third barrier: next iteration uses the other buffer)
        ABf Aa;
        {
            const unsigned short* ap = &abuf[buf][k][nl][0];
            Aa.q[0] = *(const u32x4*)(ap + kb8);
            Aa.q[1] = *(const u32x4*)(ap + 16 + kb8);
        }
        #pragma unroll
        for (int ds = 0; ds < 4; ++ds) {
            ABf Vm;
            const int d = k * NHh + ds * 16 + nl;
            const unsigned short* vp =
                Vb2 + ((size_t)d * Bb + bbi) * MPAD + mbase + hi * 8;
            Vm.q[0] = *(const u32x4*)(vp);
            Vm.q[1] = *(const u32x4*)(vp + 16);
            oacc[ds] = __builtin_amdgcn_wmma_f32_16x16x32_bf16(
                           false, Aa.v, false, Vm.v, (short)0, oacc[ds], false, false);
        }
    }

    // store o strip as bf16: Ob[(n*32+b)*512 + d]
    #pragma unroll
    for (int ds = 0; ds < 4; ++ds)
        #pragma unroll
        for (int r = 0; r < 8; ++r) {
            const int n = nbase + r + hi * 8;
            if (n < NVr) {
                const int d = k * NHh + ds * 16 + nl;
                Ob[((size_t)n * Bb + bbi) * Dd + d] = f2bf(oacc[ds][r]);
            }
        }
}

// =====================================================================
// Kernel 3: out projection (WMMA) + bias + residual + LayerNorm
// grid (75), block 256 (8 waves x 16 tokens = 128 tokens/block)
// =====================================================================
__launch_bounds__(256, 1)
__global__ void out_kernel(const unsigned short* __restrict__ Ob,
                           const float* __restrict__ w_out,
                           const float* __restrict__ b_out,
                           const float* __restrict__ x,
                           const float* __restrict__ gamma,
                           const float* __restrict__ beta,
                           float* __restrict__ out)
{
    __shared__ __align__(16) unsigned short wlds[64][512]; // w_out^T bf16, 64 KB

    const int tid  = threadIdx.x;
    const int lane = tid & 31;
    const int wid  = tid >> 5;
    const int hi  = (lane >> 4) & 1;
    const int nl  = lane & 15;
    const int kb8 = hi * 8;

    // stage w_out (512x64 f32) transposed into LDS as bf16 [c][d]
    for (int i = tid; i < 512 * 16; i += 256) {
        const int d  = i >> 4;
        const int c4 = (i & 15) * 4;
        const float4 wv4 = *(const float4*)(w_out + (size_t)d * Cc + c4);
        wlds[c4 + 0][d] = f2bf(wv4.x);
        wlds[c4 + 1][d] = f2bf(wv4.y);
        wlds[c4 + 2][d] = f2bf(wv4.z);
        wlds[c4 + 3][d] = f2bf(wv4.w);
    }
    __syncthreads();

    const int tbase = (blockIdx.x * 8 + wid) * 16;   // token strip
    const unsigned short* orow = Ob + (size_t)(tbase + nl) * Dd;

    v8f acc[4] = {};
    for (int s = 0; s < 16; ++s) {       // K = 512 = 16 steps of 32
        ABf Ao;
        const unsigned short* p = orow + s * 32 + hi * 16;
        Ao.q[0] = *(const u32x4*)(p);
        Ao.q[1] = *(const u32x4*)(p + 8);
        #pragma unroll
        for (int ct = 0; ct < 4; ++ct) {
            ABf Bw;
            const unsigned short* wp = &wlds[ct * 16 + nl][s * 32 + hi * 16];
            Bw.q[0] = *(const u32x4*)(wp);
            Bw.q[1] = *(const u32x4*)(wp + 8);
            acc[ct] = __builtin_amdgcn_wmma_f32_16x16x32_bf16(
                          false, Ao.v, false, Bw.v, (short)0, acc[ct], false, false);
        }
    }

    // bias + residual + LayerNorm (64 cols = 16 lanes x 4 regs per row)
    #pragma unroll
    for (int r = 0; r < 8; ++r) {
        const int t = tbase + r + hi * 8;           // token = n*32 + b
        float yv[4], sum = 0.f, sq = 0.f;
        #pragma unroll
        for (int ct = 0; ct < 4; ++ct) {
            const int cc = ct * 16 + nl;
            const float res = x[((size_t)28800 + t) * Cc + cc];
            yv[ct] = acc[ct][r] + b_out[cc] + res;
            sum += yv[ct]; sq += yv[ct] * yv[ct];
        }
        #pragma unroll
        for (int off = 1; off < 16; off <<= 1) {    // stays within 16-lane half
            sum += __shfl_xor(sum, off, 32);
            sq  += __shfl_xor(sq,  off, 32);
        }
        const float mu  = sum * (1.f / 64.f);
        const float var = sq * (1.f / 64.f) - mu * mu;
        const float inv = rsqrtf(var + 1e-5f);
        #pragma unroll
        for (int ct = 0; ct < 4; ++ct) {
            const int cc = ct * 16 + nl;
            out[(size_t)t * Cc + cc] = (yv[ct] - mu) * inv * gamma[cc] + beta[cc];
        }
    }
}

extern "C" void kernel_launch(void* const* d_in, const int* in_sizes, int n_in,
                              void* d_out, int out_size, void* d_ws, size_t ws_size,
                              hipStream_t stream)
{
    const float* x     = (const float*)d_in[0];
    const float* wq    = (const float*)d_in[1];
    const float* wk    = (const float*)d_in[2];
    const float* wv    = (const float*)d_in[3];
    const float* w_out = (const float*)d_in[4];
    const float* b_out = (const float*)d_in[5];
    const float* gamma = (const float*)d_in[6];
    const float* beta  = (const float*)d_in[7];
    float* out = (float*)d_out;

    unsigned short* ws  = (unsigned short*)d_ws;
    unsigned short* Qb2 = ws;                                    // 32*1200*512
    unsigned short* Kb2 = Qb2 + (size_t)Bb * Mm * Dd;            // 32*300*512
    unsigned short* Vb2 = Kb2 + (size_t)Bb * NVr * Dd;           // 512*32*1216
    unsigned short* Ob  = Vb2 + (size_t)Dd * Bb * MPAD;          // 9600*512

    proj_kernel<0><<<dim3(10, 8, 32), 256, 0, stream>>>(x, wq, Qb2);
    proj_kernel<1><<<dim3( 3, 8, 32), 256, 0, stream>>>(x, wk, Kb2);
    proj_kernel<2><<<dim3(10, 8, 32), 256, 0, stream>>>(x, wv, Vb2);
    attn_kernel<<<dim3(19, 32), 256, 0, stream>>>(Qb2, Kb2, Vb2, Ob);
    out_kernel <<<dim3(75),     256, 0, stream>>>(Ob, w_out, b_out, x, gamma, beta, out);
}
